// CoDeformableDetrTransformerDecoder_65051574665327
// MI455X (gfx1250) — compile-verified, matrix operand order checked by reading, not run on hardware
//
#include <hip/hip_runtime.h>

// ---------------------------------------------------------------------------
// CDNA5 (gfx1250) wave32 WMMA implementation of a 6-layer Deformable-DETR
// decoder.  All GEMMs run through v_wmma_f32_16x16x32_bf16 (bf16 in, f32
// accumulate).  Weights are converted to bf16 once; the value projection is
// hoisted out of the layer loop.  Attention is a fused flash-style kernel
// (online softmax, QK^T and P*V both WMMA).  MSDA bilinear sampling is a
// per-(b,q,h)-wave gather kernel (lane = channel).
// ---------------------------------------------------------------------------

typedef __attribute__((ext_vector_type(16))) __bf16 bf16x16;
typedef __attribute__((ext_vector_type(8)))  __bf16 bf16x8;
typedef __attribute__((ext_vector_type(8)))  float  f32x8;

union UA { bf16x16 v; __bf16 e[16]; };
union UC { f32x8  v; float  e[8]; };

#define WMMA_BF16(A, B, C) \
  __builtin_amdgcn_wmma_f32_16x16x32_bf16(false, (A), false, (B), (short)0, (C), false, false)

#define BB 8
#define QQ 900
#define DD 256
#define SS 13294
#define BQ (BB * QQ)          // 7200
#define QT 57                 // ceil(900/16)
#define EPSF 1e-5f
#define SCALEF 0.17677669529663687f   // 1/sqrt(32)

// ---------------------------------------------------------------------------
// f32 -> bf16 weight conversion (run once per launch, weights stay in L2)
// ---------------------------------------------------------------------------
__global__ void __launch_bounds__(256) cvt_bf16_k(const float* __restrict__ w,
                                                  __bf16* __restrict__ o, int n) {
  int i = blockIdx.x * 256 + threadIdx.x;
  if (i < n) o[i] = (__bf16)w[i];
}

// ---------------------------------------------------------------------------
// GEMM: C[M,N] = A[M,K] (f32, converted in-kernel) * W[N,K]^T (bf16) + bias
// EPI: 0=none, 1=+residual, 2=ReLU.  One wave = 16x64 output tile, 8 waves/WG.
// M % 16 == 0, N % 64 == 0, K % 32 == 0.
// ---------------------------------------------------------------------------
template <int EPI>
__global__ void __launch_bounds__(256) gemm_bf16w(
    const float* __restrict__ A, const __bf16* __restrict__ W,
    const float* __restrict__ bias, const float* __restrict__ R,
    float* __restrict__ C, int M, int N, int K) {
  const int lane = threadIdx.x & 31;
  const int wave = threadIdx.x >> 5;
  const int nt = N >> 6;
  const int tiles = (M >> 4) * nt;
  const int tile = blockIdx.x * 8 + wave;
  if (tile >= tiles) return;
  const int tm = tile / nt, tn = tile % nt;
  const int half = lane >> 4, l16 = lane & 15;

  UC acc[4];
#pragma unroll
  for (int t = 0; t < 4; ++t)
#pragma unroll
    for (int i = 0; i < 8; ++i) acc[t].e[i] = 0.0f;

  const float* arow = A + (size_t)(tm * 16 + l16) * K;
  for (int kb = 0; kb < K; kb += 32) {
    UA a;
    {
      float tmp[16];
      const float* p = arow + kb + 8 * half;
      *(float4*)(tmp + 0)  = *(const float4*)(p + 0);
      *(float4*)(tmp + 4)  = *(const float4*)(p + 4);
      *(float4*)(tmp + 8)  = *(const float4*)(p + 16);
      *(float4*)(tmp + 12) = *(const float4*)(p + 20);
#pragma unroll
      for (int i = 0; i < 16; ++i) a.e[i] = (__bf16)tmp[i];
    }
#pragma unroll
    for (int t = 0; t < 4; ++t) {
      UA bm;
      const __bf16* wp = W + (size_t)(tn * 64 + t * 16 + l16) * K + kb + 8 * half;
      *(bf16x8*)(bm.e + 0) = *(const bf16x8*)(wp + 0);
      *(bf16x8*)(bm.e + 8) = *(const bf16x8*)(wp + 16);
      acc[t].v = WMMA_BF16(a.v, bm.v, acc[t].v);
    }
  }
#pragma unroll
  for (int t = 0; t < 4; ++t) {
    const int cn = tn * 64 + t * 16 + l16;
    const float bv = bias[cn];
#pragma unroll
    for (int i = 0; i < 8; ++i) {
      const int rm = tm * 16 + i + 8 * half;
      float v = acc[t].e[i] + bv;
      if (EPI == 1) v += R[(size_t)rm * N + cn];
      if (EPI == 2) v = fmaxf(v, 0.0f);
      C[(size_t)rm * N + cn] = v;
    }
  }
}

// ---------------------------------------------------------------------------
// Fused flash attention.  qkv layout (B, Q, 768): q | k | v, head-major 32.
// One wave (32 threads) per (b, h, 16-query tile); online softmax over all
// 900 keys in 32-key WMMA blocks.  P goes C-layout -> A-layout via LDS.
// ---------------------------------------------------------------------------
__global__ void __launch_bounds__(32) attn_flash(const float* __restrict__ qkv,
                                                 float* __restrict__ out) {
  __shared__ __bf16 pl[16 * 32];
  const int lane = threadIdx.x;
  const int t = blockIdx.x;
  const int qt = t % QT;
  const int h = (t / QT) & 7;
  const int b = t / (QT * 8);
  const int half = lane >> 4, l16 = lane & 15;
  const int q0 = qt * 16;

  UA qa;
  {
    int qrow = q0 + l16; if (qrow > QQ - 1) qrow = QQ - 1;
    const float* qp = qkv + (size_t)(b * QQ + qrow) * 768 + h * 32 + 8 * half;
    float tmp[16];
    *(float4*)(tmp + 0)  = *(const float4*)(qp + 0);
    *(float4*)(tmp + 4)  = *(const float4*)(qp + 4);
    *(float4*)(tmp + 8)  = *(const float4*)(qp + 16);
    *(float4*)(tmp + 12) = *(const float4*)(qp + 20);
#pragma unroll
    for (int i = 0; i < 16; ++i) qa.e[i] = (__bf16)tmp[i];
  }

  float mrun[8], lrun[8];
  UC acc0, acc1, z;
#pragma unroll
  for (int i = 0; i < 8; ++i) {
    mrun[i] = -1e30f; lrun[i] = 0.0f;
    acc0.e[i] = 0.0f; acc1.e[i] = 0.0f; z.e[i] = 0.0f;
  }

  for (int kb = 0; kb < QQ; kb += 32) {
    UA k0, k1;
    {
      int key = kb + l16; if (key > QQ - 1) key = QQ - 1;
      const float* kp = qkv + (size_t)(b * QQ + key) * 768 + 256 + h * 32 + 8 * half;
      float tmp[16];
      *(float4*)(tmp + 0)  = *(const float4*)(kp + 0);
      *(float4*)(tmp + 4)  = *(const float4*)(kp + 4);
      *(float4*)(tmp + 8)  = *(const float4*)(kp + 16);
      *(float4*)(tmp + 12) = *(const float4*)(kp + 20);
#pragma unroll
      for (int i = 0; i < 16; ++i) k0.e[i] = (__bf16)tmp[i];
      key = kb + 16 + l16; if (key > QQ - 1) key = QQ - 1;
      kp = qkv + (size_t)(b * QQ + key) * 768 + 256 + h * 32 + 8 * half;
      *(float4*)(tmp + 0)  = *(const float4*)(kp + 0);
      *(float4*)(tmp + 4)  = *(const float4*)(kp + 4);
      *(float4*)(tmp + 8)  = *(const float4*)(kp + 16);
      *(float4*)(tmp + 12) = *(const float4*)(kp + 20);
#pragma unroll
      for (int i = 0; i < 16; ++i) k1.e[i] = (__bf16)tmp[i];
    }
    UC s0, s1;
    s0.v = WMMA_BF16(qa.v, k0.v, z.v);
    s1.v = WMMA_BF16(qa.v, k1.v, z.v);
#pragma unroll
    for (int i = 0; i < 8; ++i) {
      float a0 = s0.e[i] * SCALEF; if (kb + l16 >= QQ) a0 = -1e30f;
      float a1 = s1.e[i] * SCALEF; if (kb + 16 + l16 >= QQ) a1 = -1e30f;
      s0.e[i] = a0; s1.e[i] = a1;
    }
#pragma unroll
    for (int i = 0; i < 8; ++i) {
      float mb = fmaxf(s0.e[i], s1.e[i]);
#pragma unroll
      for (int d = 1; d < 16; d <<= 1) mb = fmaxf(mb, __shfl_xor(mb, d, 32));
      float mnew = fmaxf(mrun[i], mb);
      float alpha = __expf(mrun[i] - mnew);
      lrun[i] *= alpha; acc0.e[i] *= alpha; acc1.e[i] *= alpha;
      mrun[i] = mnew;
      float p0 = __expf(s0.e[i] - mnew);
      float p1 = __expf(s1.e[i] - mnew);
      float rs = p0 + p1;
#pragma unroll
      for (int d = 1; d < 16; d <<= 1) rs += __shfl_xor(rs, d, 32);
      lrun[i] += rs;
      pl[(i + 8 * half) * 32 + l16] = (__bf16)p0;
      pl[(i + 8 * half) * 32 + 16 + l16] = (__bf16)p1;
    }
    __builtin_amdgcn_wave_barrier();
    UA pa;
    *(bf16x8*)(pa.e + 0) = *(const bf16x8*)(pl + l16 * 32 + 8 * half);
    *(bf16x8*)(pa.e + 8) = *(const bf16x8*)(pl + l16 * 32 + 8 * half + 16);
    UA v0, v1;
#pragma unroll
    for (int e = 0; e < 8; ++e) {
      int key = kb + 8 * half + e; if (key > QQ - 1) key = QQ - 1;
      const float* vp = qkv + (size_t)(b * QQ + key) * 768 + 512 + h * 32;
      v0.e[e] = (__bf16)vp[l16];
      v1.e[e] = (__bf16)vp[16 + l16];
      key = kb + 8 * half + 16 + e; if (key > QQ - 1) key = QQ - 1;
      vp = qkv + (size_t)(b * QQ + key) * 768 + 512 + h * 32;
      v0.e[8 + e] = (__bf16)vp[l16];
      v1.e[8 + e] = (__bf16)vp[16 + l16];
    }
    acc0.v = WMMA_BF16(pa.v, v0.v, acc0.v);
    acc1.v = WMMA_BF16(pa.v, v1.v, acc1.v);
  }
#pragma unroll
  for (int i = 0; i < 8; ++i) {
    const int q = q0 + i + 8 * half;
    if (q < QQ) {
      const float inv = 1.0f / lrun[i];
      out[(size_t)(b * QQ + q) * 256 + h * 32 + l16] = acc0.e[i] * inv;
      out[(size_t)(b * QQ + q) * 256 + h * 32 + 16 + l16] = acc1.e[i] * inv;
    }
  }
}

// ---------------------------------------------------------------------------
// LayerNorm over 256-dim rows.  One wave per row, 8 rows per workgroup.
// ---------------------------------------------------------------------------
__global__ void __launch_bounds__(256) layernorm_k(
    const float* __restrict__ x, const float* __restrict__ g,
    const float* __restrict__ bt, float* __restrict__ o, int rows) {
  const int lane = threadIdx.x & 31;
  const int row = blockIdx.x * 8 + (threadIdx.x >> 5);
  if (row >= rows) return;
  const float* xr = x + (size_t)row * 256;
  float v[8], s = 0.0f, sq = 0.0f;
#pragma unroll
  for (int j = 0; j < 8; ++j) {
    v[j] = xr[lane + 32 * j];
    s += v[j]; sq += v[j] * v[j];
  }
#pragma unroll
  for (int d = 1; d < 32; d <<= 1) {
    s += __shfl_xor(s, d, 32);
    sq += __shfl_xor(sq, d, 32);
  }
  const float mu = s * (1.0f / 256.0f);
  const float var = sq * (1.0f / 256.0f) - mu * mu;
  const float r = rsqrtf(var + EPSF);
  float* orow = o + (size_t)row * 256;
#pragma unroll
  for (int j = 0; j < 8; ++j) {
    const int c = lane + 32 * j;
    orow[c] = (v[j] - mu) * r * g[c] + bt[c];
  }
}

// ---------------------------------------------------------------------------
// MSDA core: one wave per (b,q,h); lane = channel (DH=32).
// off (B,Q,256): strides h:32, lvl:8, p:2.  awr (B,Q,128): h:16.
// value (B,S,256).  refin (B,Q,4,2).
// ---------------------------------------------------------------------------
__global__ void __launch_bounds__(256) msda_k(
    const float* __restrict__ value, const float* __restrict__ off,
    const float* __restrict__ awr, const float* __restrict__ refin,
    float* __restrict__ out) {
  const int lane = threadIdx.x & 31;
  const int gid = blockIdx.x * 8 + (threadIdx.x >> 5);
  const int h = gid & 7;
  const int q = (gid >> 3) % QQ;
  const int b = gid / (8 * QQ);

  const float* ap = awr + (size_t)(b * QQ + q) * 128 + h * 16;
  float aval[16], am = -1e30f;
#pragma unroll
  for (int j = 0; j < 16; ++j) { aval[j] = ap[j]; am = fmaxf(am, aval[j]); }
  float asum = 0.0f;
#pragma unroll
  for (int j = 0; j < 16; ++j) { aval[j] = __expf(aval[j] - am); asum += aval[j]; }
  const float ainv = 1.0f / asum;

  const int HW[4] = {100, 50, 25, 13};
  const int ST[4] = {0, 10000, 12500, 13125};
  const float* op = off + (size_t)(b * QQ + q) * 256 + h * 32;
  const float* rp = refin + (size_t)(b * QQ + q) * 8;

  float acc = 0.0f;
#pragma unroll
  for (int lvl = 0; lvl < 4; ++lvl) {
    const int W = HW[lvl], H = HW[lvl];
    const float fw = (float)W, fh = (float)H;
    const float rx = rp[lvl * 2], ry = rp[lvl * 2 + 1];
    const float* vb = value + ((size_t)b * SS + ST[lvl]) * 256 + h * 32 + lane;
#pragma unroll
    for (int p = 0; p < 4; ++p) {
      const float lx = rx + op[lvl * 8 + p * 2] / fw;
      const float ly = ry + op[lvl * 8 + p * 2 + 1] / fh;
      const float xf = lx * fw - 0.5f, yf = ly * fh - 0.5f;
      const float x0f = floorf(xf), y0f = floorf(yf);
      const float wx = xf - x0f, wy = yf - y0f;
      const int x0 = (int)x0f, y0 = (int)y0f;
      float g = 0.0f;
#pragma unroll
      for (int c = 0; c < 4; ++c) {
        const int dy = c >> 1, dx = c & 1;
        const int xi = x0 + dx, yi = y0 + dy;
        const float w = (dx ? wx : 1.0f - wx) * (dy ? wy : 1.0f - wy);
        const bool ok = (xi >= 0) & (xi < W) & (yi >= 0) & (yi < H);
        const int xc = xi < 0 ? 0 : (xi > W - 1 ? W - 1 : xi);
        const int yc = yi < 0 ? 0 : (yi > H - 1 ? H - 1 : yi);
        const float vv = vb[(size_t)(yc * W + xc) * 256];
        g += ok ? vv * w : 0.0f;
      }
      acc += aval[lvl * 4 + p] * ainv * g;
    }
  }
  out[(size_t)(b * QQ + q) * 256 + h * 32 + lane] = acc;
}

// ---------------------------------------------------------------------------
// Elementwise helpers
// ---------------------------------------------------------------------------
__global__ void __launch_bounds__(256) add_k(const float* __restrict__ a,
                                             const float* __restrict__ b,
                                             float* __restrict__ o, int n) {
  int i = blockIdx.x * 256 + threadIdx.x;
  if (i < n) o[i] = a[i] + b[i];
}
__global__ void __launch_bounds__(256) copy_k(const float* __restrict__ a,
                                              float* __restrict__ o, int n) {
  int i = blockIdx.x * 256 + threadIdx.x;
  if (i < n) o[i] = a[i];
}
// refin[b,q,lvl,c] = ref_pts[b,q,c] * valid_ratios[b,lvl,c]
__global__ void __launch_bounds__(256) refin_k(const float* __restrict__ refp,
                                               const float* __restrict__ vr,
                                               float* __restrict__ o) {
  int i = blockIdx.x * 256 + threadIdx.x;
  if (i >= BQ * 8) return;
  const int c = i & 1, lvl = (i >> 1) & 3, bq = i >> 3, b = bq / QQ;
  o[i] = refp[bq * 2 + c] * vr[(b * 4 + lvl) * 2 + c];
}
// inter_refs: NLAYERS copies of reference_points
__global__ void __launch_bounds__(256) refs_k(const float* __restrict__ refp,
                                              float* __restrict__ o) {
  int i = blockIdx.x * 256 + threadIdx.x;
  if (i < 6 * BQ * 2) o[i] = refp[i % (BQ * 2)];
}

// ---------------------------------------------------------------------------
// Host launcher
// ---------------------------------------------------------------------------
static inline int gemm_grid(int M, int N) { return ((M / 16) * (N / 64) + 7) / 8; }
#define NBLK(n) (((n) + 255) / 256)

extern "C" void kernel_launch(void* const* d_in, const int* in_sizes, int n_in,
                              void* d_out, int out_size, void* d_ws, size_t ws_size,
                              hipStream_t stream) {
  (void)in_sizes; (void)n_in; (void)out_size; (void)ws_size;

  const float* tgt      = (const float*)d_in[0];
  const float* refp     = (const float*)d_in[1];
  const float* memory   = (const float*)d_in[2];
  const float* vr       = (const float*)d_in[5];
  const float* qpos     = (const float*)d_in[6];
  const float* sa_in_w  = (const float*)d_in[7];
  const float* sa_in_b  = (const float*)d_in[8];
  const float* sa_out_w = (const float*)d_in[9];
  const float* sa_out_b = (const float*)d_in[10];
  const float* n1_g = (const float*)d_in[11];
  const float* n1_b = (const float*)d_in[12];
  const float* n2_g = (const float*)d_in[13];
  const float* n2_b = (const float*)d_in[14];
  const float* n3_g = (const float*)d_in[15];
  const float* n3_b = (const float*)d_in[16];
  const float* vp_w = (const float*)d_in[17];
  const float* vp_b = (const float*)d_in[18];
  const float* so_w = (const float*)d_in[19];
  const float* so_b = (const float*)d_in[20];
  const float* aw_w = (const float*)d_in[21];
  const float* aw_b = (const float*)d_in[22];
  const float* op_w = (const float*)d_in[23];
  const float* op_b = (const float*)d_in[24];
  const float* ff1_w = (const float*)d_in[25];
  const float* ff1_b = (const float*)d_in[26];
  const float* ff2_w = (const float*)d_in[27];
  const float* ff2_b = (const float*)d_in[28];

  float* outp = (float*)d_out;                 // [6, B, Q, 256] then [6, B, Q, 2]
  float* out_refs = outp + 6 * (size_t)BQ * 256;

  // ---- workspace carve (256B aligned) ----
  char* ws = (char*)d_ws;
  size_t cur = 0;
  auto carve = [&](size_t bytes) -> char* {
    char* p = ws + cur;
    cur += (bytes + 255) & ~(size_t)255;
    return p;
  };
  float* value = (float*)carve((size_t)BB * SS * 256 * 4);
  float* x1    = (float*)carve((size_t)BQ * 256 * 4);
  float* qkv   = (float*)carve((size_t)BQ * 768 * 4);
  float* attn  = (float*)carve((size_t)BQ * 256 * 4);
  float* pre   = (float*)carve((size_t)BQ * 256 * 4);
  float* outb  = (float*)carve((size_t)BQ * 256 * 4);
  float* offb  = (float*)carve((size_t)BQ * 256 * 4);
  float* awr   = (float*)carve((size_t)BQ * 128 * 4);
  float* msd   = (float*)carve((size_t)BQ * 256 * 4);
  float* ffh   = (float*)carve((size_t)BQ * 1024 * 4);
  float* refin = (float*)carve((size_t)BQ * 8 * 4);
  __bf16* b_sa_in  = (__bf16*)carve(768 * 256 * 2);
  __bf16* b_sa_out = (__bf16*)carve(256 * 256 * 2);
  __bf16* b_vp     = (__bf16*)carve(256 * 256 * 2);
  __bf16* b_so     = (__bf16*)carve(256 * 256 * 2);
  __bf16* b_aw     = (__bf16*)carve(128 * 256 * 2);
  __bf16* b_op     = (__bf16*)carve(256 * 256 * 2);
  __bf16* b_ff1    = (__bf16*)carve(1024 * 256 * 2);
  __bf16* b_ff2    = (__bf16*)carve(256 * 1024 * 2);

  // ---- one-time prep ----
  cvt_bf16_k<<<NBLK(768 * 256), 256, 0, stream>>>(sa_in_w, b_sa_in, 768 * 256);
  cvt_bf16_k<<<NBLK(256 * 256), 256, 0, stream>>>(sa_out_w, b_sa_out, 256 * 256);
  cvt_bf16_k<<<NBLK(256 * 256), 256, 0, stream>>>(vp_w, b_vp, 256 * 256);
  cvt_bf16_k<<<NBLK(256 * 256), 256, 0, stream>>>(so_w, b_so, 256 * 256);
  cvt_bf16_k<<<NBLK(128 * 256), 256, 0, stream>>>(aw_w, b_aw, 128 * 256);
  cvt_bf16_k<<<NBLK(256 * 256), 256, 0, stream>>>(op_w, b_op, 256 * 256);
  cvt_bf16_k<<<NBLK(1024 * 256), 256, 0, stream>>>(ff1_w, b_ff1, 1024 * 256);
  cvt_bf16_k<<<NBLK(256 * 1024), 256, 0, stream>>>(ff2_w, b_ff2, 256 * 1024);

  refin_k<<<NBLK(BQ * 8), 256, 0, stream>>>(refp, vr, refin);
  refs_k<<<NBLK(6 * BQ * 2), 256, 0, stream>>>(refp, out_refs);
  copy_k<<<NBLK(BQ * 256), 256, 0, stream>>>(tgt, outb, BQ * 256);

  // value projection (hoisted out of the layer loop): (B*S,256) @ vp_w^T
  gemm_bf16w<0><<<gemm_grid(BB * SS, 256), 256, 0, stream>>>(
      memory, b_vp, vp_b, nullptr, value, BB * SS, 256, 256);

  // ---- 6 decoder layers ----
  for (int layer = 0; layer < 6; ++layer) {
    // x1 = output + query_pos
    add_k<<<NBLK(BQ * 256), 256, 0, stream>>>(outb, qpos, x1, BQ * 256);
    // qkv
    gemm_bf16w<0><<<gemm_grid(BQ, 768), 256, 0, stream>>>(
        x1, b_sa_in, sa_in_b, nullptr, qkv, BQ, 768, 256);
    // fused attention
    attn_flash<<<BB * 8 * QT, 32, 0, stream>>>(qkv, attn);
    // out proj + residual(x1)
    gemm_bf16w<1><<<gemm_grid(BQ, 256), 256, 0, stream>>>(
        attn, b_sa_out, sa_out_b, x1, pre, BQ, 256, 256);
    layernorm_k<<<BQ / 8, 256, 0, stream>>>(pre, n1_g, n1_b, outb, BQ);
    // sampling offsets + attention weights
    gemm_bf16w<0><<<gemm_grid(BQ, 256), 256, 0, stream>>>(
        outb, b_so, so_b, nullptr, offb, BQ, 256, 256);
    gemm_bf16w<0><<<gemm_grid(BQ, 128), 256, 0, stream>>>(
        outb, b_aw, aw_b, nullptr, awr, BQ, 128, 256);
    // deformable sampling
    msda_k<<<(BB * QQ * 8) / 8, 256, 0, stream>>>(value, offb, awr, refin, msd);
    // output proj + residual(outb)
    gemm_bf16w<1><<<gemm_grid(BQ, 256), 256, 0, stream>>>(
        msd, b_op, op_b, outb, pre, BQ, 256, 256);
    layernorm_k<<<BQ / 8, 256, 0, stream>>>(pre, n2_g, n2_b, outb, BQ);
    // FFN
    gemm_bf16w<2><<<gemm_grid(BQ, 1024), 256, 0, stream>>>(
        outb, b_ff1, ff1_b, nullptr, ffh, BQ, 1024, 256);
    gemm_bf16w<1><<<gemm_grid(BQ, 256), 256, 0, stream>>>(
        ffh, b_ff2, ff2_b, outb, pre, BQ, 256, 1024);
    layernorm_k<<<BQ / 8, 256, 0, stream>>>(pre, n3_g, n3_b, outb, BQ);
    // stash layer output
    copy_k<<<NBLK(BQ * 256), 256, 0, stream>>>(
        outb, outp + (size_t)layer * BQ * 256, BQ * 256);
  }
}
